// Attention_29291676959343
// MI455X (gfx1250) — matrix operand order repaired and checked by reading, not compiled, not run on hardware
//
#include <hip/hip_runtime.h>

#define DEV_INLINE __device__ __forceinline__

typedef __attribute__((ext_vector_type(16))) __bf16 v16bf;
typedef __attribute__((ext_vector_type(8)))  float  v8f;
typedef __attribute__((ext_vector_type(4)))  unsigned int v4u;
typedef __attribute__((ext_vector_type(8)))  int    v8i;
typedef __attribute__((ext_vector_type(4)))  int    v4i;

namespace {
enum : int { kB = 2, kS = 2048, kD = 2048, kH = 16, kHD = 128,
             kM = kB * kS, k3D = 3 * kD, kQKV = k3D };
constexpr float kScale = 0.08838834764831845f; // 1/sqrt(128)
constexpr float kNegInf = -1e30f;

union FragBF16 { v16bf v; uint4 q[2]; };

DEV_INLINE v8f vzero8() { v8f z; for (int i = 0; i < 8; ++i) z[i] = 0.0f; return z; }

// A-matrix fragment (16x32, 16-bit). ISA layout: lanes 0-15 hold row M=lane with
// K={0..7,16..23}; lanes 16-31 hold row M=lane-16 with K={8..15,24..31}.
DEV_INLINE v16bf load_frag_a(const __bf16* base, int stride, int lane) {
  const int row = lane & 15;
  const int kb  = (lane >> 4) * 8;
  FragBF16 f;
  f.q[0] = *(const uint4*)(base + row * stride + kb);
  f.q[1] = *(const uint4*)(base + row * stride + kb + 16);
  return f.v;
}

// B-matrix fragment (32x16, 16-bit), source stored NT (K-contiguous per column):
// lanes 0-15: col N=lane, K={0..7,16..23} ; lanes 16-31: col N=lane-16, K={8..15,24..31}.
DEV_INLINE v16bf load_frag_b(const __bf16* base, int stride, int lane) {
  const int col = lane & 15;
  const int kb  = (lane >> 4) * 8;
  FragBF16 f;
  f.q[0] = *(const uint4*)(base + col * stride + kb);
  f.q[1] = *(const uint4*)(base + col * stride + kb + 16);
  return f.v;
}

DEV_INLINE v8f wmma_bf16(v16bf a, v16bf b, v8f c) {
  return __builtin_amdgcn_wmma_f32_16x16x32_bf16(
      /*neg_a=*/false, a, /*neg_b=*/false, b,
      /*c_mod=*/(short)0, c, /*reuse_a=*/false, /*reuse_b=*/false);
}

DEV_INLINE unsigned lds_off(const void* p) {
  // Generic-address low 32 bits == LDS byte offset (flat aperture mapping).
  return (unsigned)(size_t)p;
}

// Issue a TDM 2D tile load: tile_d1 rows of tile_d0 bf16 elements, source rows
// strided by row_stride elements, packed contiguously into LDS at lds_addr.
// D# layout per CDNA5 ISA (group0: count/lds/global/type,
// group1: data_size=2B, tensor dims == tile dims (no OOB clipping needed),
// dim0 stride). Tracked by TENSORcnt. 6-arg builtin form (clang-23 signature).
DEV_INLINE void tdm_load_2d(unsigned lds_addr, const void* gptr,
                            int tile_d0, int tile_d1, long long row_stride) {
  const unsigned long long ga = (unsigned long long)(size_t)gptr;
  v4u g0;
  g0[0] = 1u;                                   // count=1, user descriptor
  g0[1] = lds_addr;                             // LDS byte address
  g0[2] = (unsigned)ga;                         // global_addr[31:0]
  g0[3] = ((unsigned)(ga >> 32) & 0x01ffffffu)  // global_addr[56:32]
          | 0x80000000u;                        // type=2 ("image")
  const unsigned ud0 = (unsigned)tile_d0, ud1 = (unsigned)tile_d1;
  v8i g1;
  g1[0] = (int)(1u << 16);                      // data_size=1 (2 bytes); no multicast/pad
  g1[1] = (int)((ud0 & 0xffffu) << 16);         // tensor_dim0[15:0]
  g1[2] = (int)(((ud0 >> 16) & 0xffffu) | ((ud1 & 0xffffu) << 16)); // dim0 hi | dim1 lo
  g1[3] = (int)(((ud1 >> 16) & 0xffffu) | ((ud0 & 0xffffu) << 16)); // dim1 hi | tile_dim0
  g1[4] = (int)(ud1 & 0xffffu);                 // tile_dim1 (tile_dim2 = 0)
  g1[5] = (int)(unsigned)(row_stride & 0xffffffffLL);       // dim0_stride[31:0]
  g1[6] = (int)(unsigned)((row_stride >> 32) & 0xffffLL);   // dim0_stride[47:32]
  g1[7] = 0;
  const v4i z4 = {0, 0, 0, 0};
  const v8i z8 = {0, 0, 0, 0, 0, 0, 0, 0};
  __builtin_amdgcn_tensor_load_to_lds(g0, g1, z4, z4, z8, 0);
}
} // namespace

// ---------------------------------------------------------------------------
// fp32 -> bf16 conversion
// ---------------------------------------------------------------------------
__global__ void cvt_f32_to_bf16(const float* __restrict__ in,
                                __bf16* __restrict__ out, int n) {
  int i = blockIdx.x * blockDim.x + threadIdx.x;
  int stride = gridDim.x * blockDim.x;
  for (; i < n; i += stride) out[i] = (__bf16)in[i];
}

// ---------------------------------------------------------------------------
// NT GEMM: C[M,N] = A[M,K] * B[N,K]^T, bf16 in / fp32 accumulate.
// Block tile 128x128, 8 waves, each computing a 32x64 sub-tile (2x4 WMMA
// fragments). Tiles are streamed into double-buffered LDS by the Tensor Data
// Mover (wave 0 issues, TENSORcnt-synchronized), overlapping DMA of step t+1
// with WMMA compute of step t.
// ---------------------------------------------------------------------------
template <typename OutT>
__global__ __launch_bounds__(256) void gemm_bf16_nt(
    const __bf16* __restrict__ A, const __bf16* __restrict__ B,
    OutT* __restrict__ C, int M, int N, int K) {
  __shared__ __attribute__((aligned(16))) __bf16 As[2][128 * 32];
  __shared__ __attribute__((aligned(16))) __bf16 Bs[2][128 * 32];

  const int m0 = blockIdx.y * 128;
  const int n0 = blockIdx.x * 128;
  const int tid = threadIdx.x;
  const int lane = tid & 31;
  const int w = tid >> 5;
  const int wm = (w >> 1) * 32;
  const int wn = (w & 1) * 64;

  v8f acc[2][4];
  for (int i = 0; i < 2; ++i)
    for (int j = 0; j < 4; ++j) acc[i][j] = vzero8();

  const int nsteps = K >> 5;
  if (w == 0) {  // prologue: DMA first tiles into buffer 0
    tdm_load_2d(lds_off(&As[0][0]), A + (size_t)m0 * K, 32, 128, K);
    tdm_load_2d(lds_off(&Bs[0][0]), B + (size_t)n0 * K, 32, 128, K);
  }

  for (int t = 0; t < nsteps; ++t) {
    const int buf = t & 1;
    if (w == 0) {
      if (t + 1 < nsteps) {  // prefetch next tiles into the other buffer
        const int k1 = (t + 1) << 5;
        tdm_load_2d(lds_off(&As[buf ^ 1][0]), A + (size_t)m0 * K + k1, 32, 128, K);
        tdm_load_2d(lds_off(&Bs[buf ^ 1][0]), B + (size_t)n0 * K + k1, 32, 128, K);
        __builtin_amdgcn_s_wait_tensorcnt(2);  // current pair landed; next in flight
      } else {
        __builtin_amdgcn_s_wait_tensorcnt(0);
      }
    }
    __syncthreads();  // release current buffer to all waves

    v16bf af[2], bfr[4];
    for (int i = 0; i < 2; ++i)
      af[i] = load_frag_a(&As[buf][(wm + 16 * i) * 32], 32, lane);
    for (int j = 0; j < 4; ++j)
      bfr[j] = load_frag_b(&Bs[buf][(wn + 16 * j) * 32], 32, lane);
    for (int i = 0; i < 2; ++i)
      for (int j = 0; j < 4; ++j)
        acc[i][j] = wmma_bf16(af[i], bfr[j], acc[i][j]);
    __syncthreads();  // compute done before TDM overwrites this buffer
  }

  // C/D layout: lane = (half<<4)|nn : row = r + 8*half, col = nn.
  const int half = lane >> 4;
  const int cc = lane & 15;
  for (int i = 0; i < 2; ++i) {
    for (int j = 0; j < 4; ++j) {
      const size_t rbase = (size_t)(m0 + wm + 16 * i + 8 * half);
      const int col = n0 + wn + 16 * j + cc;
      for (int r = 0; r < 8; ++r)
        C[(rbase + r) * (size_t)N + col] = (OutT)acc[i][j][r];
    }
  }
}

// ---------------------------------------------------------------------------
// RoPE applied in-place to q and k halves of the bf16 qkv buffer.
// ---------------------------------------------------------------------------
__global__ __launch_bounds__(256) void rope_kernel(
    __bf16* __restrict__ qkv, const float* __restrict__ fcos,
    const float* __restrict__ fsin) {
  const int idx = blockIdx.x * blockDim.x + threadIdx.x; // < B*S*H*64
  const int p = idx & 63;
  const int h = (idx >> 6) & 15;
  const int s = (idx >> 10) & 2047;
  const int b = idx >> 21;
  const float c = fcos[s * 64 + p];
  const float sn = fsin[s * 64 + p];
  const size_t row = (size_t)(b * kS + s) * kQKV;
  for (int qk = 0; qk < 2; ++qk) {
    const size_t off = row + (size_t)qk * kD + h * kHD + 2 * p;
    const float re = (float)qkv[off];
    const float im = (float)qkv[off + 1];
    qkv[off]     = (__bf16)(re * c - im * sn);
    qkv[off + 1] = (__bf16)(re * sn + im * c);
  }
}

// ---------------------------------------------------------------------------
// Flash attention (causal, online softmax). One workgroup = one (b,h) and a
// 64-query tile; key loop in 32-key steps. Q and K tiles arrive via TDM;
// V is loaded manually because it is transposed into LDS for the B operand.
// ---------------------------------------------------------------------------
__global__ __launch_bounds__(256) void flash_attn_kernel(
    const __bf16* __restrict__ qkv, __bf16* __restrict__ ctx) {
  __shared__ __attribute__((aligned(16))) __bf16 Qs[64 * 128]; // 16 KB
  __shared__ __attribute__((aligned(16))) __bf16 Ks[32 * 128]; //  8 KB
  __shared__ __attribute__((aligned(16))) __bf16 Vt[128 * 32]; //  8 KB (transposed)
  __shared__ __attribute__((aligned(16))) float  Ss[64 * 32];  //  8 KB
  __shared__ __attribute__((aligned(16))) __bf16 Ps[64 * 32];  //  4 KB
  __shared__ float mrow[64], lrow[64], arow[64];

  const int q0 = blockIdx.x * 64;
  const int bh = blockIdx.y;
  const int b = bh / kH;
  const int h = bh % kH;
  const int tid = threadIdx.x;
  const int lane = tid & 31;
  const int w = tid >> 5;

  const __bf16* Qg = qkv + (size_t)(b * kS) * kQKV + h * kHD;
  const __bf16* Kg = Qg + kD;
  const __bf16* Vg = Qg + 2 * kD;

  if (w == 0)  // DMA Q tile (64 rows x 128) once
    tdm_load_2d(lds_off(&Qs[0]), Qg + (size_t)q0 * kQKV, 128, 64, kQKV);
  if (tid < 64) { mrow[tid] = kNegInf; lrow[tid] = 0.0f; }

  const int oi = w >> 1;
  const int oj0 = (w & 1) * 4;
  v8f o[4];
  for (int j = 0; j < 4; ++j) o[j] = vzero8();

  const int half = lane >> 4;
  const int nn = lane & 15;
  const int nkt = (q0 >> 5) + 2; // causal: keys up to q0+63

  for (int kt = 0; kt < nkt; ++kt) {
    const int k0 = kt * 32;
    if (w == 0)  // DMA K tile (32 x 128)
      tdm_load_2d(lds_off(&Ks[0]), Kg + (size_t)k0 * kQKV, 128, 32, kQKV);
    // Manual transposed V load: Vt[hd][key].
    for (int c = tid; c < 512; c += 256) {
      const int r = c >> 4;            // key index 0..31
      const int kc = (c & 15) * 8;     // hd chunk
      uint4 vv = *(const uint4*)(Vg + (size_t)(k0 + r) * kQKV + kc);
      const __bf16* pv = (const __bf16*)&vv;
      for (int t = 0; t < 8; ++t) Vt[(kc + t) * 32 + r] = pv[t];
    }
    if (w == 0) __builtin_amdgcn_s_wait_tensorcnt(0);
    __syncthreads();

    // S = Q * K^T : 64x32 = 4x2 tiles, one per wave.
    const int si = w >> 1;
    const int sj = w & 1;
    v8f s = vzero8();
    for (int ks = 0; ks < 128; ks += 32) {
      v16bf aq = load_frag_a(&Qs[(si * 16) * 128 + ks], 128, lane);
      v16bf bk = load_frag_b(&Ks[(sj * 16) * 128 + ks], 128, lane);
      s = wmma_bf16(aq, bk, s);
    }
    for (int r = 0; r < 8; ++r)
      Ss[(si * 16 + r + 8 * half) * 32 + sj * 16 + nn] = s[r];
    __syncthreads();

    // Online softmax: thread t < 64 owns query row t.
    if (tid < 64) {
      const int qi = q0 + tid;
      const float mold = mrow[tid];
      float mx = mold;
      for (int n = 0; n < 32; ++n) {
        const int ki = k0 + n;
        float v = (ki <= qi) ? Ss[tid * 32 + n] * kScale : kNegInf;
        Ss[tid * 32 + n] = v;
        mx = fmaxf(mx, v);
      }
      const float alpha = __expf(mold - mx);
      float sum = 0.0f;
      for (int n = 0; n < 32; ++n) {
        const float p = __expf(Ss[tid * 32 + n] - mx);
        Ps[tid * 32 + n] = (__bf16)p;
        sum += p;
      }
      mrow[tid] = mx;
      arow[tid] = alpha;
      lrow[tid] = lrow[tid] * alpha + sum;
    }
    __syncthreads();

    // Rescale accumulators, then O += P * V  (Kdim = 32, single step).
    float asc[8];
    for (int r = 0; r < 8; ++r) asc[r] = arow[16 * oi + r + 8 * half];
    for (int j = 0; j < 4; ++j)
      for (int r = 0; r < 8; ++r) o[j][r] *= asc[r];

    v16bf ap = load_frag_a(&Ps[(oi * 16) * 32], 32, lane);
    for (int j = 0; j < 4; ++j) {
      v16bf bv = load_frag_b(&Vt[((oj0 + j) * 16) * 32], 32, lane);
      o[j] = wmma_bf16(ap, bv, o[j]);
    }
    __syncthreads(); // protect Ks/Vt/Ss/Ps for the next key tile
  }

  // Epilogue: normalize by row sums, write context (bf16) at [b, s, h, hd].
  for (int j = 0; j < 4; ++j) {
    for (int r = 0; r < 8; ++r) {
      const int row = 16 * oi + r + 8 * half;
      const int col = (oj0 + j) * 16 + nn;
      const float val = o[j][r] / lrow[row];
      ctx[(size_t)(b * kS + q0 + row) * kD + h * kHD + col] = (__bf16)val;
    }
  }
}

// ---------------------------------------------------------------------------
// Host launcher
// ---------------------------------------------------------------------------
extern "C" void kernel_launch(void* const* d_in, const int* in_sizes, int n_in,
                              void* d_out, int out_size, void* d_ws, size_t ws_size,
                              hipStream_t stream) {
  const float* x    = (const float*)d_in[0];
  const float* wqkv = (const float*)d_in[1];
  const float* wout = (const float*)d_in[2];
  const float* fcos = (const float*)d_in[3];
  const float* fsin = (const float*)d_in[4];
  float* out = (float*)d_out;

  char* ws = (char*)d_ws;
  size_t off = 0;
  __bf16* xh    = (__bf16*)(ws + off); off += (size_t)kM  * kD  * sizeof(__bf16);
  __bf16* wqkvh = (__bf16*)(ws + off); off += (size_t)k3D * kD  * sizeof(__bf16);
  __bf16* wouth = (__bf16*)(ws + off); off += (size_t)kD  * kD  * sizeof(__bf16);
  __bf16* qkvb  = (__bf16*)(ws + off); off += (size_t)kM  * k3D * sizeof(__bf16);
  __bf16* ctx   = (__bf16*)(ws + off); off += (size_t)kM  * kD  * sizeof(__bf16);
  (void)ws_size; (void)in_sizes; (void)n_in; (void)out_size;

  const int n_x  = kM * kD;        // 8.4M
  const int n_wq = k3D * kD;       // 12.6M
  const int n_wo = kD * kD;        // 4.2M
  cvt_f32_to_bf16<<<(n_x  + 255) / 256, 256, 0, stream>>>(x,    xh,    n_x);
  cvt_f32_to_bf16<<<(n_wq + 255) / 256, 256, 0, stream>>>(wqkv, wqkvh, n_wq);
  cvt_f32_to_bf16<<<(n_wo + 255) / 256, 256, 0, stream>>>(wout, wouth, n_wo);

  // QKV projection: [4096,6144] = xh[4096,2048] * wqkvh[6144,2048]^T
  gemm_bf16_nt<__bf16><<<dim3(k3D / 128, kM / 128), 256, 0, stream>>>(
      xh, wqkvh, qkvb, kM, k3D, kD);

  // RoPE on q and k halves.
  rope_kernel<<<(kB * kS * kH * (kHD / 2)) / 256, 256, 0, stream>>>(
      qkvb, fcos, fsin);

  // Causal flash attention -> context [4096, 2048] bf16.
  flash_attn_kernel<<<dim3(kS / 64, kB * kH), 256, 0, stream>>>(qkvb, ctx);

  // Output projection: out[4096,2048] = ctx * wouth^T (fp32 result).
  gemm_bf16_nt<float><<<dim3(kD / 128, kM / 128), 256, 0, stream>>>(
      ctx, wouth, out, kM, kD, kD);
}
